// DenseMoE_10574209483533
// MI455X (gfx1250) — matrix operand order repaired and checked by reading, not compiled, not run on hardware
//
#include <hip/hip_runtime.h>
#include <hip/hip_bf16.h>

// MI455X (gfx1250) dense-MoE: bf16 WMMA + TDM (tensor_load_to_lds) / async-LDS double buffering.
// Sizes fixed by the reference: T=4096, D=2048, F=3584, E=8, top_k=2.

typedef __attribute__((ext_vector_type(16))) __bf16 bf16x16;
typedef __attribute__((ext_vector_type(8)))  __bf16 bf16x8;
typedef __attribute__((ext_vector_type(8)))  float  f32x8;

#define TOK 4096
#define DIM 2048
#define FFN 3584
#define NEXP 8
#define LDS_PITCH 40   // bf16 elements per LDS row (32 data + 8 pad -> 80B pitch, bank-conflict free)

#if defined(__gfx1250__) && __has_builtin(__builtin_amdgcn_global_load_async_to_lds_b128) && \
    __has_builtin(__builtin_amdgcn_s_wait_asynccnt)
#define HAVE_ASYNC_LDS 1
#else
#define HAVE_ASYNC_LDS 0
#endif

#if defined(__gfx1250__) && __has_builtin(__builtin_amdgcn_tensor_load_to_lds) && \
    __has_builtin(__builtin_amdgcn_s_wait_tensorcnt)
#define HAVE_TDM 1
#else
#define HAVE_TDM 0
#endif

#if HAVE_ASYNC_LDS
#define ASYNC_WAIT(n) __builtin_amdgcn_s_wait_asynccnt(n)
#else
#define ASYNC_WAIT(n) ((void)0)
#endif

#if HAVE_TDM
#define TENSOR_WAIT(n) __builtin_amdgcn_s_wait_tensorcnt(n)
#else
#define TENSOR_WAIT(n) ((void)0)
#endif

// exact parameter types of the async-LDS builtin (from clang's diagnostic):
// pointer to GCC-vector int4, global (AS1) source / LDS (AS3) destination
typedef int v4i_gcc __attribute__((vector_size(16)));
typedef unsigned int v4u_gcc __attribute__((vector_size(16)));
typedef int v8i_gcc __attribute__((vector_size(32)));
typedef __attribute__((address_space(1))) v4i_gcc* as1_v4i_ptr;
typedef __attribute__((address_space(3))) v4i_gcc* as3_v4i_ptr;

// copy 16 bytes global -> LDS (async path bypasses VGPRs, tracked by ASYNCcnt)
__device__ __forceinline__ void cp16_g2l(const unsigned short* g, unsigned short* l) {
#if HAVE_ASYNC_LDS
  __builtin_amdgcn_global_load_async_to_lds_b128((as1_v4i_ptr)g, (as3_v4i_ptr)l, 0, 0);
#else
  *reinterpret_cast<uint4*>(l) = *reinterpret_cast<const uint4*>(g);
#endif
}

// byte offset of a __shared__ object within the workgroup's LDS
__device__ __forceinline__ unsigned int lds_byte_off(const unsigned short* p) {
  return (unsigned int)(uintptr_t)(__attribute__((address_space(3))) const unsigned short*)p;
}

#if HAVE_TDM
// TDM: DMA a 128-row x 32-element bf16 tile (row stride = stride_elems) from global
// into LDS at lds_off, inserting 4 DWORDs of padding after every 16 DWORDs so the
// LDS image has the kernel's 80-byte row pitch. D# layout per CDNA5 ISA ch.8.
// This toolchain's builtin is the 6-arg form:
//   (uint32x4 g0, int32x8 g1, int32x4 g2, int32x4 g3, int32x8 extra, i32 cpol)
__device__ __forceinline__ void tdm_load_tile(const unsigned short* g, unsigned int lds_off,
                                              unsigned int stride_elems) {
  unsigned long long ga = (unsigned long long)(uintptr_t)g;
  v4u_gcc g0 = {};
  g0[0] = 1u;                                             // count=1 valid descriptor
  g0[1] = lds_off;                                        // lds_addr (bytes)
  g0[2] = (unsigned int)(ga & 0xFFFFFFFFu);               // global_addr[31:0]
  g0[3] = (unsigned int)((ga >> 32) & 0x01FFFFFFu)        // global_addr[56:32]
        | (2u << 30);                                     // type=2 ("image")
  v8i_gcc g1 = {};
  g1[0] = (int)((1u << 16)                                // data_size = 2 bytes
              | (1u << 20)                                // pad_enable
              | (3u << 22)                                // pad_interval: 16 DWORDs
              | (3u << 25));                              // pad_amount: 4 DWORDs
  g1[1] = (int)((stride_elems & 0xFFFFu) << 16);          // tensor_dim0[15:0] (row length)
  g1[2] = (int)((stride_elems >> 16) | (0x4000u << 16));  // tensor_dim0 hi | tensor_dim1 lo (16384 rows)
  g1[3] = (int)(32u << 16);                               // tensor_dim1 hi | tile_dim0 = 32
  g1[4] = (int)128u;                                      // tile_dim1 = 128 rows, tile_dim2 = 0
  g1[5] = (int)stride_elems;                              // tensor_dim0_stride lo32
  g1[6] = 0;                                              // stride hi | tensor_dim1_stride lo
  g1[7] = 0;
  v4i_gcc z4 = {};
  v8i_gcc z8 = {};
  __builtin_amdgcn_tensor_load_to_lds(g0, g1, z4, z4, z8, 0);
}
#endif

__device__ __forceinline__ unsigned short f32_bf16(float f) {
  union { float f; unsigned u; } v; v.f = f;
  unsigned r = v.u + 0x7FFFu + ((v.u >> 16) & 1u);   // round-to-nearest-even
  return (unsigned short)(r >> 16);
}

// A fragment (16x32 bf16): lane m<16 holds row m K[0..7]+K[16..23]; lane m+16 holds K[8..15]+K[24..31]
__device__ __forceinline__ bf16x16 lds_fragA(const unsigned short* s, int row0, int lane) {
  int r = row0 + (lane & 15);
  int k = (lane >> 4) << 3;
  bf16x8 c0 = *reinterpret_cast<const bf16x8*>(s + r * LDS_PITCH + k);
  bf16x8 c1 = *reinterpret_cast<const bf16x8*>(s + r * LDS_PITCH + k + 16);
  return __builtin_shufflevector(c0, c1, 0,1,2,3,4,5,6,7,8,9,10,11,12,13,14,15);
}

// B fragment (32x16 bf16, column-per-lane): lane n<16 holds col n K[0..7]+K[16..23]; lane n+16 holds col n K[8..15]+K[24..31]
__device__ __forceinline__ bf16x16 lds_fragB(const unsigned short* s, int col0, int lane) {
  int r = col0 + (lane & 15);
  int k = (lane >> 4) << 3;
  bf16x8 c0 = *reinterpret_cast<const bf16x8*>(s + r * LDS_PITCH + k);
  bf16x8 c1 = *reinterpret_cast<const bf16x8*>(s + r * LDS_PITCH + k + 16);
  return __builtin_shufflevector(c0, c1, 0,1,2,3,4,5,6,7,8,9,10,11,12,13,14,15);
}

__device__ __forceinline__ f32x8 wmma_bf16(bf16x16 a, bf16x16 b, f32x8 c) {
  return __builtin_amdgcn_wmma_f32_16x16x32_bf16(false, a, false, b, (short)0, c, false, false);
}

// ---------------- gating: logits -> top-2 -> renormalized weights ----------------
__global__ __launch_bounds__(128) void moe_gate_kernel(const float* __restrict__ x,
                                                       const float* __restrict__ Wg,
                                                       float* __restrict__ wgate) {
  __shared__ float sWg[NEXP * DIM];
  for (int i = threadIdx.x; i < (NEXP * DIM) / 4; i += 128)
    reinterpret_cast<float4*>(sWg)[i] = reinterpret_cast<const float4*>(Wg)[i];
  __syncthreads();

  int t = blockIdx.x * 128 + threadIdx.x;
  const float* xr = x + (size_t)t * DIM;
  float acc[NEXP];
#pragma unroll
  for (int e = 0; e < NEXP; ++e) acc[e] = 0.f;
  for (int d = 0; d < DIM; d += 4) {
    float4 xv = *reinterpret_cast<const float4*>(xr + d);
#pragma unroll
    for (int e = 0; e < NEXP; ++e) {
      const float* wr = sWg + e * DIM + d;
      acc[e] += xv.x * wr[0] + xv.y * wr[1] + xv.z * wr[2] + xv.w * wr[3];
    }
  }
  int i1 = 0;
#pragma unroll
  for (int e = 1; e < NEXP; ++e) if (acc[e] > acc[i1]) i1 = e;
  int i2 = (i1 == 0) ? 1 : 0;
#pragma unroll
  for (int e = 0; e < NEXP; ++e) if (e != i1 && acc[e] > acc[i2]) i2 = e;
  // softmax normalization cancels under top-2 L1 renorm: w = exp(l-l1)/(1+exp(l2-l1))
  float p2 = __expf(acc[i2] - acc[i1]);
  float s = 1.f + p2;
#pragma unroll
  for (int e = 0; e < NEXP; ++e)
    wgate[(size_t)t * NEXP + e] = (e == i1) ? (1.f / s) : ((e == i2) ? (p2 / s) : 0.f);
}

// ---------------- fp32 -> bf16 conversion ----------------
__global__ __launch_bounds__(256) void cvt_bf16_kernel(const float* __restrict__ src,
                                                       unsigned short* __restrict__ dst,
                                                       long long n4) {
  long long i = (long long)blockIdx.x * 256 + threadIdx.x;
  long long stride = (long long)gridDim.x * 256;
  for (; i < n4; i += stride) {
    float4 v = reinterpret_cast<const float4*>(src)[i];
    ushort4 o;
    o.x = f32_bf16(v.x); o.y = f32_bf16(v.y); o.z = f32_bf16(v.z); o.w = f32_bf16(v.w);
    reinterpret_cast<ushort4*>(dst)[i] = o;
  }
}

// ---------------- GEMM1: H[e] = wgate[:,e] * silu(X@W1[e]^T) * (X@V1[e]^T) ----------------
// grid: (FFN/128, TOK/128, NEXP), block 256 (8 waves). Block tile 128x128, wave tile 32x64.
// Double-buffered LDS; wave 0 drives the TDM for next-tile copies (fallback: per-lane async).
__global__ void __launch_bounds__(256) moe_gemm1_kernel(const unsigned short* __restrict__ Xbf,
                                                        const unsigned short* __restrict__ W1bf,
                                                        const unsigned short* __restrict__ V1bf,
                                                        const float* __restrict__ wgate,
                                                        unsigned short* __restrict__ Hbf) {
  __shared__ unsigned short sA[2][128 * LDS_PITCH];
  __shared__ unsigned short sBw[2][128 * LDS_PITCH];
  __shared__ unsigned short sBv[2][128 * LDS_PITCH];
  __shared__ float sW[128];

  const int e = blockIdx.z;
  const int nBase = blockIdx.x * 128;   // FFN dim
  const int mBase = blockIdx.y * 128;   // token dim
  const unsigned short* W1e = W1bf + (size_t)e * FFN * DIM;
  const unsigned short* V1e = V1bf + (size_t)e * FFN * DIM;

  const int tid = threadIdx.x;
  const int lane = tid & 31;
  const int wave = tid >> 5;
  const int wm = (wave & 3) * 32;   // wave M offset within block tile
  const int wn = (wave >> 2) * 64;  // wave N offset within block tile

  if (tid < 128) sW[tid] = wgate[(size_t)(mBase + tid) * NEXP + e];

  auto issue_tile = [&](int k0, int buf) {
#if HAVE_TDM
    if (wave == 0) {     // one DMA per tile; TDM ignores EXEC, branch is wave-uniform
      tdm_load_tile(Xbf + (size_t)mBase * DIM + k0, lds_byte_off(&sA[buf][0]),  DIM);
      tdm_load_tile(W1e + (size_t)nBase * DIM + k0, lds_byte_off(&sBw[buf][0]), DIM);
      tdm_load_tile(V1e + (size_t)nBase * DIM + k0, lds_byte_off(&sBv[buf][0]), DIM);
    }
#else
#pragma unroll
    for (int j = 0; j < 2; ++j) {
      int lin = tid * 2 + j;          // 0..511 -> 128 rows x 4 x(8 bf16)
      int r = lin >> 2, c = (lin & 3) * 8;
      cp16_g2l(Xbf + (size_t)(mBase + r) * DIM + k0 + c, &sA[buf][r * LDS_PITCH + c]);
      cp16_g2l(W1e + (size_t)(nBase + r) * DIM + k0 + c, &sBw[buf][r * LDS_PITCH + c]);
      cp16_g2l(V1e + (size_t)(nBase + r) * DIM + k0 + c, &sBv[buf][r * LDS_PITCH + c]);
    }
#endif
  };

  f32x8 accP[2][4] = {};
  f32x8 accQ[2][4] = {};

  issue_tile(0, 0);
  for (int k0 = 0; k0 < DIM; k0 += 32) {
    const int cur = (k0 >> 5) & 1;
    __syncthreads();                       // all waves done *reading* buf cur^1 (prev iter)
    if (k0 + 32 < DIM) {
      issue_tile(k0 + 32, cur ^ 1);        // overlap next-tile DMA with this tile's WMMAs
#if HAVE_TDM
      if (wave == 0) TENSOR_WAIT(3);       // retire tile k's 3 DMAs (in-order per wave)
#else
      ASYNC_WAIT(6);                       // retire tile k copies (in-order completion)
#endif
    } else {
#if HAVE_TDM
      if (wave == 0) TENSOR_WAIT(0);
#else
      ASYNC_WAIT(0);
#endif
    }
    __syncthreads();                       // tile k visible to all waves

    bf16x16 aF[2];
#pragma unroll
    for (int i = 0; i < 2; ++i) aF[i] = lds_fragA(sA[cur], wm + i * 16, lane);
#pragma unroll
    for (int j = 0; j < 4; ++j) {
      bf16x16 bw = lds_fragB(sBw[cur], wn + j * 16, lane);
      bf16x16 bv = lds_fragB(sBv[cur], wn + j * 16, lane);
#pragma unroll
      for (int i = 0; i < 2; ++i) {
        accP[i][j] = wmma_bf16(aF[i], bw, accP[i][j]);
        accQ[i][j] = wmma_bf16(aF[i], bv, accQ[i][j]);
      }
    }
  }

  // epilogue: h = w[t,e] * silu(p) * q, store bf16 into H[e][t][f]
#pragma unroll
  for (int i = 0; i < 2; ++i) {
#pragma unroll
    for (int j = 0; j < 4; ++j) {
#pragma unroll
      for (int v = 0; v < 8; ++v) {
        int row = wm + i * 16 + v + ((lane >> 4) << 3);   // local M (C layout: vgpr v -> M=v or v+8)
        int col = nBase + wn + j * 16 + (lane & 15);      // global F
        float p = accP[i][j][v];
        float q = accQ[i][j][v];
        float sig = 1.f / (1.f + __expf(-p));
        float h = sW[row] * (p * sig) * q;
        Hbf[(size_t)e * TOK * FFN + (size_t)(mBase + row) * FFN + col] = f32_bf16(h);
      }
    }
  }
}

// ---------------- GEMM2: out = sum_e H[e] @ W2[e]^T ----------------
// grid: (DIM/128, TOK/128), block 256. Block tile 128x128, wave tile 32x64, K = NEXP*FFN.
__global__ void __launch_bounds__(256) moe_gemm2_kernel(const unsigned short* __restrict__ Hbf,
                                                        const unsigned short* __restrict__ W2bf,
                                                        float* __restrict__ out) {
  __shared__ unsigned short sA[2][128 * LDS_PITCH];
  __shared__ unsigned short sB[2][128 * LDS_PITCH];

  const int nBase = blockIdx.x * 128;   // DIM
  const int mBase = blockIdx.y * 128;   // token dim
  const int tid = threadIdx.x;
  const int lane = tid & 31;
  const int wave = tid >> 5;
  const int wm = (wave & 3) * 32;
  const int wn = (wave >> 2) * 64;

  const int KT = FFN / 32;              // K-tiles per expert
  const int NT = NEXP * KT;             // total K-tiles

  auto issue_tile = [&](int t, int buf) {
    int ee = t / KT;
    int k0 = (t - ee * KT) * 32;
    const unsigned short* Ae = Hbf  + (size_t)ee * TOK * FFN;
    const unsigned short* Be = W2bf + (size_t)ee * DIM * FFN;
#if HAVE_TDM
    if (wave == 0) {
      tdm_load_tile(Ae + (size_t)mBase * FFN + k0, lds_byte_off(&sA[buf][0]), FFN);
      tdm_load_tile(Be + (size_t)nBase * FFN + k0, lds_byte_off(&sB[buf][0]), FFN);
    }
#else
#pragma unroll
    for (int j = 0; j < 2; ++j) {
      int lin = tid * 2 + j;
      int r = lin >> 2, c = (lin & 3) * 8;
      cp16_g2l(Ae + (size_t)(mBase + r) * FFN + k0 + c, &sA[buf][r * LDS_PITCH + c]);
      cp16_g2l(Be + (size_t)(nBase + r) * FFN + k0 + c, &sB[buf][r * LDS_PITCH + c]);
    }
#endif
  };

  f32x8 acc[2][4] = {};

  issue_tile(0, 0);
  for (int t = 0; t < NT; ++t) {
    const int cur = t & 1;
    __syncthreads();
    if (t + 1 < NT) {
      issue_tile(t + 1, cur ^ 1);
#if HAVE_TDM
      if (wave == 0) TENSOR_WAIT(2);
#else
      ASYNC_WAIT(4);
#endif
    } else {
#if HAVE_TDM
      if (wave == 0) TENSOR_WAIT(0);
#else
      ASYNC_WAIT(0);
#endif
    }
    __syncthreads();

    bf16x16 aF[2];
#pragma unroll
    for (int i = 0; i < 2; ++i) aF[i] = lds_fragA(sA[cur], wm + i * 16, lane);
#pragma unroll
    for (int j = 0; j < 4; ++j) {
      bf16x16 b = lds_fragB(sB[cur], wn + j * 16, lane);
#pragma unroll
      for (int i = 0; i < 2; ++i) acc[i][j] = wmma_bf16(aF[i], b, acc[i][j]);
    }
  }

#pragma unroll
  for (int i = 0; i < 2; ++i)
#pragma unroll
    for (int j = 0; j < 4; ++j)
#pragma unroll
      for (int v = 0; v < 8; ++v) {
        int row = wm + i * 16 + v + ((lane >> 4) << 3);
        int col = nBase + wn + j * 16 + (lane & 15);
        out[(size_t)(mBase + row) * DIM + col] = acc[i][j][v];
      }
}

extern "C" void kernel_launch(void* const* d_in, const int* in_sizes, int n_in,
                              void* d_out, int out_size, void* d_ws, size_t ws_size,
                              hipStream_t stream) {
  (void)in_sizes; (void)n_in; (void)out_size; (void)ws_size;
  const float* x  = (const float*)d_in[0];
  const float* Wg = (const float*)d_in[1];
  const float* W1 = (const float*)d_in[2];
  const float* V1 = (const float*)d_in[3];
  const float* W2 = (const float*)d_in[4];
  float* out = (float*)d_out;

  // workspace layout
  char* ws = (char*)d_ws;
  const size_t nX  = (size_t)TOK * DIM;          // 8.4M
  const size_t nW  = (size_t)NEXP * FFN * DIM;   // 58.7M (W1, V1, W2)
  const size_t nH  = (size_t)NEXP * TOK * FFN;   // 117.4M
  float*          wgate = (float*)ws;            ws += (size_t)TOK * NEXP * sizeof(float);
  unsigned short* xbf   = (unsigned short*)ws;   ws += nX * 2;
  unsigned short* w1bf  = (unsigned short*)ws;   ws += nW * 2;
  unsigned short* v1bf  = (unsigned short*)ws;   ws += nW * 2;
  unsigned short* w2bf  = (unsigned short*)ws;   ws += nW * 2;
  unsigned short* hbf   = (unsigned short*)ws;   ws += nH * 2;

  moe_gate_kernel<<<TOK / 128, 128, 0, stream>>>(x, Wg, wgate);

  cvt_bf16_kernel<<<4096, 256, 0, stream>>>(x,  xbf,  (long long)(nX / 4));
  cvt_bf16_kernel<<<8192, 256, 0, stream>>>(W1, w1bf, (long long)(nW / 4));
  cvt_bf16_kernel<<<8192, 256, 0, stream>>>(V1, v1bf, (long long)(nW / 4));
  cvt_bf16_kernel<<<8192, 256, 0, stream>>>(W2, w2bf, (long long)(nW / 4));

  dim3 g1(FFN / 128, TOK / 128, NEXP);
  moe_gemm1_kernel<<<g1, 256, 0, stream>>>(xbf, w1bf, v1bf, wgate, hbf);

  dim3 g2(DIM / 128, TOK / 128);
  moe_gemm2_kernel<<<g2, 256, 0, stream>>>(hbf, w2bf, out);
}
